// MultiScaleTASStream_1408749273904
// MI455X (gfx1250) — matrix operand order repaired and checked by reading, not compiled, not run on hardware
//
#include <hip/hip_runtime.h>
#include <math.h>

#define B_  8
#define T_  2048
#define D_  512
#define M_  (B_*T_)        // 16384 rows

typedef __bf16 bf16;
typedef __bf16 v16bf __attribute__((ext_vector_type(16)));
typedef __bf16 v4bf  __attribute__((ext_vector_type(4)));
typedef float  v8f   __attribute__((ext_vector_type(8)));
typedef float  v4f   __attribute__((ext_vector_type(4)));
typedef unsigned int u32x4 __attribute__((ext_vector_type(4)));

union Frag { u32x4 u[2]; v16bf v; };   // 32 bytes = one 16-bit WMMA operand per lane

// ---------------------------------------------------------------- helpers
__device__ __forceinline__ float block_sum256(float v, float* red) {
  #pragma unroll
  for (int o = 16; o > 0; o >>= 1) v += __shfl_xor(v, o, 32);
  const int w = threadIdx.x >> 5, lane = threadIdx.x & 31;
  __syncthreads();
  if (lane == 0) red[w] = v;
  __syncthreads();
  float r = 0.f;
  #pragma unroll
  for (int i = 0; i < 8; i++) r += red[i];
  return r;
}

__device__ __forceinline__ float gelu_exact(float x) {
  return 0.5f * x * (1.f + erff(x * 0.70710678118654752f));
}

// ---------------------------------------------------------------- f32 -> bf16
__global__ void f32_to_bf16_kernel(const float* __restrict__ src,
                                   bf16* __restrict__ dst, int n) {
  int i = blockIdx.x * 256 + threadIdx.x;
  if (i < n) dst[i] = (bf16)src[i];
}

__global__ void zero_cnt_kernel(unsigned int* cnt) { *cnt = 0u; }

// ---------------------------------------------------------------- conv (k=2/8/32) + 3x LayerNorm, pack comb
// comb layout per row (bf16, width 2560): [emb | short | mid | long | memory]
__global__ __launch_bounds__(256)
void conv_ln_kernel(const float* __restrict__ emb,
                    const float* __restrict__ wS, const float* __restrict__ bS,
                    const float* __restrict__ gS, const float* __restrict__ beS,
                    const float* __restrict__ wM, const float* __restrict__ bM,
                    const float* __restrict__ gM, const float* __restrict__ beM,
                    const float* __restrict__ wL, const float* __restrict__ bL,
                    const float* __restrict__ gL, const float* __restrict__ beL,
                    bf16* __restrict__ comb) {
  const int r   = blockIdx.x;           // r = b*T + t
  const int t   = r & (T_ - 1);
  const int tid = threadIdx.x;
  __shared__ float red[8];
  float y[3][2];
  #pragma unroll
  for (int h = 0; h < 2; ++h) {
    const int d = tid + h * 256;
    float a0 = bS[d], a1 = bM[d], a2 = bL[d];
    #pragma unroll
    for (int i = 0; i < 2; i++)
      if (t - i >= 0) a0 += emb[(long)(r - i) * D_ + d] * wS[d * 2 + (1 - i)];
    #pragma unroll
    for (int i = 0; i < 8; i++)
      if (t - i >= 0) a1 += emb[(long)(r - i) * D_ + d] * wM[d * 8 + (7 - i)];
    for (int i = 0; i < 32; i++)
      if (t - i >= 0) a2 += emb[(long)(r - i) * D_ + d] * wL[d * 32 + (31 - i)];
    y[0][h] = a0; y[1][h] = a1; y[2][h] = a2;
    comb[(size_t)r * 2560 + d] = (bf16)emb[(long)r * D_ + d];  // emb slice
  }
  const float* gv[3]  = {gS, gM, gL};
  const float* bev[3] = {beS, beM, beL};
  for (int s = 0; s < 3; s++) {
    float sum = block_sum256(y[s][0] + y[s][1], red);
    float sq  = block_sum256(y[s][0] * y[s][0] + y[s][1] * y[s][1], red);
    float mu  = sum * (1.f / D_);
    float inv = rsqrtf(sq * (1.f / D_) - mu * mu + 1e-5f);
    #pragma unroll
    for (int h = 0; h < 2; ++h) {
      const int d = tid + h * 256;
      float v = (y[s][h] - mu) * inv * gv[s][d] + bev[s][d];
      comb[(size_t)r * 2560 + (s + 1) * D_ + d] = (bf16)v;
    }
  }
}

// ---------------------------------------------------------------- generic bf16 WMMA GEMM
// C[M,N] = act(A[M,K] @ W[N,K]^T + bias), block tile 128x64, wave tile 32x32.
// SPLIT: for k>=512 read A at (row-1 within sequence, k-512)  ->  [e_t, e_{t-1}].
template <int ACT_GELU, int OUT_BF16, int BETA, int SPLIT>
__global__ __launch_bounds__(256)
void wmma_gemm(const bf16* __restrict__ A, int lda,
               const bf16* __restrict__ W,      // (N,K) row-major
               const float* __restrict__ bias,
               void* __restrict__ Cout, int ldc,
               int M, int N, int K) {
  __shared__ __align__(16) bf16 As[2][128][40];   // +8 halves pad (16B aligned rows)
  __shared__ __align__(16) bf16 Ws[2][64][40];
  const int tid  = threadIdx.x;
  const long row0 = (long)blockIdx.x * 128;
  const int  col0 = blockIdx.y * 64;
  const int wave = tid >> 5, lane = tid & 31;
  const int wm = wave >> 1, wn = wave & 1;        // 4x2 wave grid
  const int lm = lane & 15, kh = lane >> 4;
  const int nk = K >> 5;

  v8f c[2][2] = {};
  u32x4 ra[2], rw;

  auto gload = [&](int kc) {
    #pragma unroll
    for (int i = 0; i < 2; i++) {
      int idx = tid + i * 256;
      long grow = row0 + (idx >> 2);
      int  gk   = kc * 32 + (idx & 3) * 8;
      if (SPLIT && gk >= 512) { gk -= 512; if ((grow & (T_ - 1)) != 0) grow -= 1; }
      ra[i] = *(const u32x4*)(A + grow * (long)lda + gk);
    }
    rw = *(const u32x4*)(W + (long)(col0 + (tid >> 2)) * K + kc * 32 + (tid & 3) * 8);
  };
  auto sstore = [&](int buf) {
    #pragma unroll
    for (int i = 0; i < 2; i++) {
      int idx = tid + i * 256;
      *(u32x4*)&As[buf][idx >> 2][(idx & 3) * 8] = ra[i];
    }
    *(u32x4*)&Ws[buf][tid >> 2][(tid & 3) * 8] = rw;
  };

  gload(0); sstore(0); __syncthreads();

  for (int kc = 0; kc < nk; ++kc) {
    const int buf = kc & 1;
    if (kc + 1 < nk) gload(kc + 1);
    if (kc + 2 < nk)
      __builtin_prefetch((const void*)(A + (row0 + (tid >> 1)) * (long)lda + ((kc + 2) << 5)), 0, 0);

    Frag a[2], b[2];
    #pragma unroll
    for (int mt = 0; mt < 2; mt++) {           // A: interleaved K layout (ISA 7.12.2)
      const int rr = wm * 32 + mt * 16 + lm;
      a[mt].u[0] = *(const u32x4*)&As[buf][rr][kh * 8];
      a[mt].u[1] = *(const u32x4*)&As[buf][rr][16 + kh * 8];
    }
    #pragma unroll
    for (int nt = 0; nt < 2; nt++) {           // B: interleaved K layout (same as A)
      const int cc = wn * 32 + nt * 16 + lm;
      b[nt].u[0] = *(const u32x4*)&Ws[buf][cc][kh * 8];
      b[nt].u[1] = *(const u32x4*)&Ws[buf][cc][16 + kh * 8];
    }
    #pragma unroll
    for (int mt = 0; mt < 2; mt++)
      #pragma unroll
      for (int nt = 0; nt < 2; nt++)
        c[mt][nt] = __builtin_amdgcn_wmma_f32_16x16x32_bf16(
            false, a[mt].v, false, b[nt].v, (short)0, c[mt][nt], false, false);

    __syncthreads();
    if (kc + 1 < nk) { sstore(buf ^ 1); __syncthreads(); }
  }

  #pragma unroll
  for (int mt = 0; mt < 2; mt++)
    #pragma unroll
    for (int nt = 0; nt < 2; nt++) {
      const int n = col0 + wn * 32 + nt * 16 + lm;
      const float bn = bias[n];
      #pragma unroll
      for (int g = 0; g < 8; g++) {            // C/D: lane=col, VGPR g + 8*(lane>>4) = row
        const long m = row0 + wm * 32 + mt * 16 + kh * 8 + g;
        float v = c[mt][nt][g] + bn;
        if (ACT_GELU) v = gelu_exact(v);
        const long off = m * (long)ldc + n;
        if (OUT_BF16) ((bf16*)Cout)[off] = (bf16)v;
        else { float* o = (float*)Cout; if (BETA) v += o[off]; o[off] = v; }
      }
    }
}

// ---------------------------------------------------------------- GRU scan: 32 blocks, weight-stationary
// Block k owns hidden dims [16k,16k+16); keeps 48 rows of W_hh in LDS and their WMMA
// B-fragments in registers for the whole T=2048 scan. A rows 8..15 are padded to zero
// branchlessly: lanes lm>=8 point at a zeroed LDS region instead of hbf. All 16 A-frags
// are loaded before a sched_barrier so they stay in distinct registers and the waitcnt
// pass can overlap LDS latency with the WMMA accumulation chain.
__global__ __launch_bounds__(128)
void gru_kernel(const bf16* __restrict__ Whh,    // (1536,512) bf16
                const float* __restrict__ bhh,   // (1536,)
                const float* __restrict__ xg,    // (16384,1536) f32 preacts, order r,z,n
                float* __restrict__ hG,          // (8,512) f32 shared state
                bf16* __restrict__ comb,         // memory slice at col 2048
                unsigned int* __restrict__ cnt) {
  __shared__ __align__(16) bf16 Wl[48][520];
  __shared__ __align__(16) bf16 hbf[8][520];
  __shared__ __align__(16) bf16 zbuf[544];       // zero source for pad rows
  __shared__ float hg[48 * 8];
  const int tid = threadIdx.x;
  const int ds  = blockIdx.x * 16;
  const int wave = tid >> 5, lane = tid & 31;
  const int lm = lane & 15, kh = lane >> 4;

  for (int i = tid; i < 544; i += 128) zbuf[i] = (bf16)0.f;

  // stage W_hh slice: rows {g*512+ds+i : g<3, i<16}
  for (int idx = tid; idx < 48 * 32; idx += 128) {
    const int lr = idx >> 5, seg = idx & 31;
    const long grow = (long)(lr >> 4) * 512 + ds + (lr & 15);
    *(u32x4*)&Wl[lr][seg * 8] = *(const u32x4*)(Whh + grow * 512 + seg * 8);
  }
  __syncthreads();

  Frag bfr[16];                                   // register-resident weights (128 VGPRs)
  if (wave < 3) {
    #pragma unroll
    for (int kc = 0; kc < 16; kc++) {
      const int cc = wave * 16 + lm;
      bfr[kc].u[0] = *(const u32x4*)&Wl[cc][kc * 32 + kh * 8];
      bfr[kc].u[1] = *(const u32x4*)&Wl[cc][kc * 32 + 16 + kh * 8];
    }
  }
  const float bcol = (wave < 3) ? bhh[wave * 512 + ds + lm] : 0.f;

  // per-lane branchless A-row base: real h row for lm<8, zero row otherwise
  const bf16* rowbase = (lm < 8) ? &hbf[lm][0] : &zbuf[0];

  // gate-phase constants for this thread
  const int gb = tid >> 4, gdl = tid & 15, gd = ds + gdl;
  const float* xbase = xg + (long)gb * T_ * 1536 + gd;
  bf16* cbase = comb + (long)gb * T_ * 2560 + 2048 + gd;
  float hold_reg = 0.f;                           // h_{t-1}[gb][gd] lives in a register

  for (int t = 0; t < T_; ++t) {
    // prefetch gate preactivations early (independent of h)
    const float xr = xbase[(long)t * 1536];
    const float xz = xbase[(long)t * 1536 + 512];
    const float xn = xbase[(long)t * 1536 + 1024];

    // stage h -> LDS bf16 (vectorized: float4 -> packed 4x bf16)
    for (int i = tid * 4; i < 8 * 512; i += 128 * 4) {
      const int b = i >> 9, d = i & 511;
      v4f hv = {0.f, 0.f, 0.f, 0.f};
      if (t != 0) hv = *(const v4f*)(hG + b * 512 + d);
      v4bf hb = { (bf16)hv.x, (bf16)hv.y, (bf16)hv.z, (bf16)hv.w };
      *(v4bf*)&hbf[b][d] = hb;
    }
    __syncthreads();

    if (wave < 3) {                               // hg = h @ Whh_slice^T
      Frag afr[16];                               // all 16 A-frags live at once:
      #pragma unroll
      for (int kc = 0; kc < 16; kc++) {           // 32 outstanding ds_load_b128
        afr[kc].u[0] = *(const u32x4*)(rowbase + kc * 32 + kh * 8);
        afr[kc].u[1] = *(const u32x4*)(rowbase + kc * 32 + 16 + kh * 8);
      }
      __builtin_amdgcn_sched_barrier(0);          // keep loads hoisted above the WMMA chain
      v8f c = {};
      #pragma unroll
      for (int kc = 0; kc < 16; kc++)             // accumulator-only dependency chain
        c = __builtin_amdgcn_wmma_f32_16x16x32_bf16(
            false, afr[kc].v, false, bfr[kc].v, (short)0, c, false, false);
      if (lane < 16) {                            // rows m=0..7 (batch) live in lanes 0-15
        #pragma unroll
        for (int g = 0; g < 8; g++)
          hg[(wave * 16 + lm) * 8 + g] = c[g] + bcol;
      }
    }
    __syncthreads();

    {                                             // gates for 8 batch x 16 dims
      const float hr = hg[gdl * 8 + gb], hz = hg[(16 + gdl) * 8 + gb], hn = hg[(32 + gdl) * 8 + gb];
      const float r = 1.f / (1.f + expf(-(xr + hr)));
      const float z = 1.f / (1.f + expf(-(xz + hz)));
      const float n = tanhf(xn + r * hn);
      const float hnew = (1.f - z) * n + z * hold_reg;
      hold_reg = hnew;
      hG[gb * 512 + gd] = hnew;
      cbase[(long)t * 2560] = (bf16)hnew;
    }

    // release h, device-wide step barrier (cluster barrier = NOP when not clustered)
    __threadfence();
    __syncthreads();
    __builtin_amdgcn_s_cluster_barrier();
    if (tid == 0) {
      atomicAdd(cnt, 1u);
      const unsigned target = (unsigned)(t + 1) * 32u;
      while (atomicAdd(cnt, 0u) < target) __builtin_amdgcn_s_sleep(1);
    }
    __syncthreads();
    __threadfence();                              // acquire before next h staging
  }
}

// ---------------------------------------------------------------- out = LN(emb + acc) in place on d_out
__global__ __launch_bounds__(256)
void final_ln_kernel(const float* __restrict__ emb,
                     const float* __restrict__ lng, const float* __restrict__ lnb,
                     float* __restrict__ out) {
  const int r = blockIdx.x, tid = threadIdx.x;
  __shared__ float red[8];
  float v[2];
  #pragma unroll
  for (int h = 0; h < 2; ++h) {
    const int d = tid + h * 256;
    v[h] = emb[(long)r * D_ + d] + out[(long)r * D_ + d];
  }
  float sum = block_sum256(v[0] + v[1], red);
  float sq  = block_sum256(v[0] * v[0] + v[1] * v[1], red);
  float mu  = sum * (1.f / D_);
  float inv = rsqrtf(sq * (1.f / D_) - mu * mu + 1e-5f);
  #pragma unroll
  for (int h = 0; h < 2; ++h) {
    const int d = tid + h * 256;
    out[(long)r * D_ + d] = (v[h] - mu) * inv * lng[d] + lnb[d];
  }
}

// ---------------------------------------------------------------- launch
extern "C" void kernel_launch(void* const* d_in, const int* in_sizes, int n_in,
                              void* d_out, int out_size, void* d_ws, size_t ws_size,
                              hipStream_t stream) {
  const float* emb   = (const float*)d_in[0];
  const float* wS    = (const float*)d_in[1];  const float* bS  = (const float*)d_in[2];
  const float* gS    = (const float*)d_in[3];  const float* beS = (const float*)d_in[4];
  const float* wM    = (const float*)d_in[5];  const float* bM  = (const float*)d_in[6];
  const float* gM    = (const float*)d_in[7];  const float* beM = (const float*)d_in[8];
  const float* wL    = (const float*)d_in[9];  const float* bL  = (const float*)d_in[10];
  const float* gL    = (const float*)d_in[11]; const float* beL = (const float*)d_in[12];
  const float* W_ih  = (const float*)d_in[13]; const float* W_hh = (const float*)d_in[14];
  const float* b_ih  = (const float*)d_in[15]; const float* b_hh = (const float*)d_in[16];
  const float* ce_w1 = (const float*)d_in[17]; const float* ce_b1 = (const float*)d_in[18];
  const float* ce_w2 = (const float*)d_in[19]; const float* ce_b2 = (const float*)d_in[20];
  const float* fu_w1 = (const float*)d_in[21]; const float* fu_b1 = (const float*)d_in[22];
  const float* fu_w2 = (const float*)d_in[23]; const float* fu_b2 = (const float*)d_in[24];
  const float* ln_g  = (const float*)d_in[25]; const float* ln_b  = (const float*)d_in[26];

  char* ws = (char*)d_ws;
  size_t off = 0;
  auto alloc = [&](size_t bytes) { void* p = ws + off; off += (bytes + 255) & ~(size_t)255; return p; };
  bf16*  comb   = (bf16*)alloc((size_t)M_ * 2560 * 2);
  float* xg     = (float*)alloc((size_t)M_ * 1536 * 4);
  bf16*  H1     = (bf16*)alloc((size_t)M_ * 1024 * 2);
  bf16*  H2     = (bf16*)alloc((size_t)M_ * 256 * 2);
  float* hG     = (float*)alloc((size_t)B_ * D_ * 4);
  bf16*  Wih_b  = (bf16*)alloc((size_t)1536 * 512 * 2);
  bf16*  Whh_b  = (bf16*)alloc((size_t)1536 * 512 * 2);
  bf16*  fw1_b  = (bf16*)alloc((size_t)1024 * 2560 * 2);
  bf16*  fw2_b  = (bf16*)alloc((size_t)512 * 1024 * 2);
  bf16*  cw1_b  = (bf16*)alloc((size_t)256 * 1024 * 2);
  bf16*  cw2_b  = (bf16*)alloc((size_t)512 * 256 * 2);
  unsigned int* cnt = (unsigned int*)alloc(64);

  auto cvt = [&](const float* s, bf16* d, int n) {
    f32_to_bf16_kernel<<<(n + 255) / 256, 256, 0, stream>>>(s, d, n);
  };
  cvt(W_ih, Wih_b, 1536 * 512);
  cvt(W_hh, Whh_b, 1536 * 512);
  cvt(fu_w1, fw1_b, 1024 * 2560);
  cvt(fu_w2, fw2_b, 512 * 1024);
  cvt(ce_w1, cw1_b, 256 * 1024);
  cvt(ce_w2, cw2_b, 512 * 256);

  conv_ln_kernel<<<M_, 256, 0, stream>>>(emb, wS, bS, gS, beS, wM, bM, gM, beM,
                                         wL, bL, gL, beL, comb);

  // xg = emb @ W_ih^T + b_ih  (f32 out)
  wmma_gemm<0,0,0,0><<<dim3(M_/128, 1536/64), 256, 0, stream>>>(
      comb, 2560, Wih_b, b_ih, xg, 1536, M_, 1536, 512);

  zero_cnt_kernel<<<1, 1, 0, stream>>>(cnt);
  gru_kernel<<<32, 128, 0, stream>>>(Whh_b, b_hh, xg, hG, comb, cnt);

  // H1 = gelu(comb @ fu_w1^T + b1)  (bf16)
  wmma_gemm<1,1,0,0><<<dim3(M_/128, 1024/64), 256, 0, stream>>>(
      comb, 2560, fw1_b, fu_b1, H1, 1024, M_, 1024, 2560);
  // H2 = gelu([e_t,e_{t-1}] @ ce_w1^T + b1)  (bf16, SPLIT mode reads shifted rows)
  wmma_gemm<1,1,0,1><<<dim3(M_/128, 256/64), 256, 0, stream>>>(
      comb, 2560, cw1_b, ce_b1, H2, 256, M_, 256, 1024);
  // d_out  = H1 @ fu_w2^T + b2
  wmma_gemm<0,0,0,0><<<dim3(M_/128, 512/64), 256, 0, stream>>>(
      H1, 1024, fw2_b, fu_b2, d_out, 512, M_, 512, 1024);
  // d_out += H2 @ ce_w2^T + b2
  wmma_gemm<0,0,1,0><<<dim3(M_/128, 512/64), 256, 0, stream>>>(
      H2, 256, cw2_b, ce_b2, d_out, 512, M_, 512, 256);

  final_ln_kernel<<<M_, 256, 0, stream>>>(emb, ln_g, ln_b, (float*)d_out);
}
